// GRU_Model_47837345743486
// MI455X (gfx1250) — compile-verified, matrix operand order checked by reading, not run on hardware
//
#include <hip/hip_runtime.h>

typedef __attribute__((ext_vector_type(16))) _Float16 v16h;
typedef __attribute__((ext_vector_type(8)))  _Float16 v8h;
typedef __attribute__((ext_vector_type(8)))  float    v8f;

#define B_   8192
#define T_   256
#define F_   64
#define H_   50
#define G3   150     // 3*H (source layout)
#define NP   192     // 3*64 gate-separated padded
#define NT   12      // N tiles of 16
#define WAVES 4
#define ROWS_PER_BLOCK (WAVES * 16)

__device__ __forceinline__ float sigmoid_f(float x) {
  return 1.0f / (1.0f + __expf(-x));
}
__device__ __forceinline__ float tanh_f(float x) {
  float e = __expf(-2.0f * x);
  return (1.0f - e) / (1.0f + e);
}

// Fragment layout in LDS: frag (kt,nt) occupies 32 lanes x 16 halves (32B/lane).
// Per-lane halves follow the CDNA5 16-bit A/B register layout:
//   K' = (h&7) + 16*(h>=8) + 8*(lane>=16), N' = lane&15
#define WFRAG(kt,nt) (*(const v16h*)(sW + ((((kt)*NT + (nt))*32 + lane) << 4)))
#define UFRAG(kt,nt) (*(const v16h*)(sU + ((((kt)*NT + (nt))*32 + lane) << 4)))

#define WMMA(a,b,c) __builtin_amdgcn_wmma_f32_16x16x32_f16(false, (a), false, (b), (short)0, (c), false, false)

__global__ __launch_bounds__(WAVES * 32)
void gru_fused(const float* __restrict__ x, const float* __restrict__ W,
               const float* __restrict__ U, const float* __restrict__ b,
               const float* __restrict__ Wd, const float* __restrict__ bd,
               float* __restrict__ out)
{
  __shared__ __align__(32) _Float16 sW[2 * NT * 512];   // 24 KB, B-frag layout
  __shared__ __align__(32) _Float16 sU[2 * NT * 512];   // 24 KB, B-frag layout
  __shared__ float sB0[NP], sB1[NP];                    // gate-separated biases
  __shared__ __align__(32) _Float16 sH[WAVES][16 * 64]; // h tiles, row-major [16][64]
  __shared__ float sP[WAVES][16][16];                   // final dot partials

  const int tid = threadIdx.x;

  // ---- Phase A: repack W,U into WMMA B-fragment layout, gate-separated + padded ----
  for (int e = tid; e < 2 * NT * 512; e += blockDim.x) {
    int fi   = e >> 9;            // fragment index
    int lane = (e >> 4) & 31;
    int h    = e & 15;
    int kt = fi / NT, nt = fi % NT;
    int Kp = (h & 7) + ((h >> 3) << 4) + ((lane >> 4) << 3);
    int K  = kt * 32 + Kp;                    // input-feature / hidden index
    int n  = nt * 16 + (lane & 15);           // padded gate column
    int g  = n >> 6, jj = n & 63;
    int col = g * H_ + jj;                    // source column in [0,150)
    float wv = (jj < H_) ? W[K * G3 + col] : 0.0f;             // F=64 exact in K
    float uv = (jj < H_ && K < H_) ? U[K * G3 + col] : 0.0f;   // H pad 50..63 -> 0
    sW[e] = (_Float16)wv;
    sU[e] = (_Float16)uv;
  }
  for (int n = tid; n < NP; n += blockDim.x) {
    int g = n >> 6, jj = n & 63;
    sB0[n] = (jj < H_) ? b[g * H_ + jj] : 0.0f;        // input bias b[0]
    sB1[n] = (jj < H_) ? b[G3 + g * H_ + jj] : 0.0f;   // recurrent bias b[1]
  }
  for (int e = tid; e < WAVES * 16 * 64; e += blockDim.x)
    ((_Float16*)sH)[e] = (_Float16)0.0f;               // h0 = 0
  __syncthreads();

  // ---- Phase B: per-wave GRU scan over a 16-row batch tile ----
  const int wave  = tid >> 5;
  const int lane  = tid & 31;
  const int mrow  = lane & 15;
  const int upper = lane >> 4;
  const int rowBase = blockIdx.x * ROWS_PER_BLOCK + wave * 16;
  const float* xrow = x + (size_t)(rowBase + mrow) * T_ * F_;
  _Float16* hbase = &sH[wave][0];

  // Loop-invariant per-lane bias constants (col within gate = i*16 + mrow)
  float cz[4], cr[4], bh0v[4], bh1v[4];
  #pragma unroll
  for (int i = 0; i < 4; ++i) {
    int n = i * 16 + mrow;
    cz[i]   = sB0[n]       + sB1[n];        // z gate: xp + rp biases combine
    cr[i]   = sB0[64 + n]  + sB1[64 + n];   // r gate
    bh0v[i] = sB0[128 + n];                 // h gate: kept separate (reset_after)
    bh1v[i] = sB1[128 + n];
  }

  v8f hC[4];                                // h master copy, C layout, f32
  #pragma unroll
  for (int i = 0; i < 4; ++i) hC[i] = (v8f){};

  for (int t = 0; t < T_; ++t) {
    // -- x A-fragments (f32 global -> f16 A layout) --
    v16h xA[2];
    #pragma unroll
    for (int kt = 0; kt < 2; ++kt) {
      const float* p = xrow + (size_t)t * F_ + kt * 32 + upper * 8;
      float4 a0 = ((const float4*)p)[0];
      float4 a1 = ((const float4*)p)[1];
      float4 a2 = ((const float4*)(p + 16))[0];
      float4 a3 = ((const float4*)(p + 16))[1];
      v16h v;
      v[0]=(_Float16)a0.x; v[1]=(_Float16)a0.y; v[2]=(_Float16)a0.z; v[3]=(_Float16)a0.w;
      v[4]=(_Float16)a1.x; v[5]=(_Float16)a1.y; v[6]=(_Float16)a1.z; v[7]=(_Float16)a1.w;
      v[8]=(_Float16)a2.x; v[9]=(_Float16)a2.y; v[10]=(_Float16)a2.z; v[11]=(_Float16)a2.w;
      v[12]=(_Float16)a3.x; v[13]=(_Float16)a3.y; v[14]=(_Float16)a3.z; v[15]=(_Float16)a3.w;
      xA[kt] = v;
    }
    if (t + 8 < T_) {
      __builtin_prefetch(xrow + (size_t)(t + 8) * F_ + upper * 32, 0, 0);
    }

    // -- h A-fragments from LDS (same-wave DS ops are in-order; no barrier) --
    v16h hA[2];
    #pragma unroll
    for (int kt = 0; kt < 2; ++kt) {
      const _Float16* hp = hbase + mrow * 64 + kt * 32 + upper * 8;
      v8h lo = *(const v8h*)hp;
      v8h hi = *(const v8h*)(hp + 16);
      hA[kt] = __builtin_shufflevector(lo, hi, 0,1,2,3,4,5,6,7,8,9,10,11,12,13,14,15);
    }

    // -- per hidden tile: 12 WMMAs + gates + h update (keeps accs at 4 v8f) --
    #pragma unroll
    for (int i = 0; i < 4; ++i) {
      v8f az = {}, ar = {}, ax = {}, ah = {};
      // z gate (nt = i): xz + rz accumulated together
      az = WMMA(xA[0], WFRAG(0, i),     az);
      az = WMMA(xA[1], WFRAG(1, i),     az);
      az = WMMA(hA[0], UFRAG(0, i),     az);
      az = WMMA(hA[1], UFRAG(1, i),     az);
      // r gate (nt = 4+i)
      ar = WMMA(xA[0], WFRAG(0, 4 + i), ar);
      ar = WMMA(xA[1], WFRAG(1, 4 + i), ar);
      ar = WMMA(hA[0], UFRAG(0, 4 + i), ar);
      ar = WMMA(hA[1], UFRAG(1, 4 + i), ar);
      // h gate (nt = 8+i): x-part and h-part kept separate (reset_after=True)
      ax = WMMA(xA[0], WFRAG(0, 8 + i), ax);
      ax = WMMA(xA[1], WFRAG(1, 8 + i), ax);
      ah = WMMA(hA[0], UFRAG(0, 8 + i), ah);
      ah = WMMA(hA[1], UFRAG(1, 8 + i), ah);

      #pragma unroll
      for (int j = 0; j < 8; ++j) {
        float z  = sigmoid_f(az[j] + cz[i]);
        float r  = sigmoid_f(ar[j] + cr[i]);
        float hh = tanh_f(ax[j] + bh0v[i] + r * (ah[j] + bh1v[i]));
        float hn = z * hC[i][j] + (1.0f - z) * hh;
        hC[i][j] = hn;
        // C layout -> row-major f16 h tile: row = j + 8*upper, col = 16*i + mrow
        hbase[(j + upper * 8) * 64 + i * 16 + mrow] = (_Float16)hn;
      }
    }
  }

  // ---- Final dense head: out[b] = h[b,:50] @ Wd + bd ----
  float wd[4];
  #pragma unroll
  for (int i = 0; i < 4; ++i) {
    int col = i * 16 + mrow;
    wd[i] = (col < H_) ? Wd[col] : 0.0f;
  }
  #pragma unroll
  for (int j = 0; j < 8; ++j) {
    float p = 0.0f;
    #pragma unroll
    for (int i = 0; i < 4; ++i) p += hC[i][j] * wd[i];
    sP[wave][j + upper * 8][mrow] = p;   // same-wave DS, in-order
  }
  if (lane < 16) {
    float s = bd[0];
    #pragma unroll
    for (int c = 0; c < 16; ++c) s += sP[wave][lane][c];
    out[rowBase + lane] = s;
  }
}

extern "C" void kernel_launch(void* const* d_in, const int* in_sizes, int n_in,
                              void* d_out, int out_size, void* d_ws, size_t ws_size,
                              hipStream_t stream) {
  const float* x  = (const float*)d_in[0];
  const float* W  = (const float*)d_in[1];
  const float* U  = (const float*)d_in[2];
  const float* b  = (const float*)d_in[3];
  const float* Wd = (const float*)d_in[4];
  const float* bd = (const float*)d_in[5];
  float* out = (float*)d_out;

  dim3 grid(B_ / ROWS_PER_BLOCK);   // 8192 / 64 = 128 blocks
  dim3 block(WAVES * 32);           // 4 wave32 waves
  hipLaunchKernelGGL(gru_fused, grid, block, 0, stream, x, W, U, b, Wd, bd, out);
}
